// NeighConsensus_17463337025578
// MI455X (gfx1250) — matrix operand order repaired and checked by reading, not compiled, not run on hardware
//
#include <hip/hip_runtime.h>

typedef __attribute__((ext_vector_type(16))) _Float16 v16h;
typedef __attribute__((ext_vector_type(8)))  _Float16 v8h;
typedef __attribute__((ext_vector_type(8)))  float    v8f;

#define NB    8
#define DIM   25
#define SP    (25*25*25*25)       /* 390625 spatial positions per batch   */
#define TOT   (NB*SP)             /* 3,125,000 output positions           */
#define NCH   16
#define NTAP  81
#define GUARD (1u<<20)            /* 1 MB guard band around act1 in d_ws  */

// ---------------------------------------------------------------------------
// zero the 256-byte "null page" at d_ws+0 (target of masked-out tap loads)
// ---------------------------------------------------------------------------
__global__ void nc_zero(float* __restrict__ p) { p[threadIdx.x] = 0.f; }

// ---------------------------------------------------------------------------
// conv1: 1 -> 16 channels, f32 VALU, writes channel-last f16 activations.
// perm!=0 reads the (d1,d2)<->(d3,d4) transposed input.
// ---------------------------------------------------------------------------
__global__ void nc_conv1(const float* __restrict__ x, const float* __restrict__ w1,
                         const float* __restrict__ b1, _Float16* __restrict__ act1,
                         int perm) {
  __shared__ float wl[NCH * NTAP];
  __shared__ float bl[NCH];
  for (int e = threadIdx.x; e < NCH * NTAP; e += blockDim.x) wl[e] = w1[e];
  if (threadIdx.x < NCH) bl[threadIdx.x] = b1[threadIdx.x];
  __syncthreads();

  int idx = blockIdx.x * blockDim.x + threadIdx.x;
  if (idx >= TOT) return;
  int b = idx / SP, f = idx - b * SP;
  int i4 = f % DIM;  int q = f / DIM;
  int i3 = q % DIM;  q /= DIM;
  int i2 = q % DIM;  int i1 = q / DIM;

  float acc[NCH];
#pragma unroll
  for (int c = 0; c < NCH; ++c) acc[c] = 0.f;

  for (int t = 0; t < NTAP; ++t) {
    int k4 = t % 3, r = t / 3;
    int k3 = r % 3; r /= 3;
    int k2 = r % 3; int k1 = r / 3;
    int j1 = i1 + k1 - 1, j2 = i2 + k2 - 1, j3 = i3 + k3 - 1, j4 = i4 + k4 - 1;
    if ((unsigned)j1 < DIM && (unsigned)j2 < DIM &&
        (unsigned)j3 < DIM && (unsigned)j4 < DIM) {
      int flat = perm ? (((j3 * DIM + j4) * DIM + j1) * DIM + j2)
                      : (((j1 * DIM + j2) * DIM + j3) * DIM + j4);
      float v = x[(size_t)b * SP + flat];
#pragma unroll
      for (int c = 0; c < NCH; ++c) acc[c] += wl[c * NTAP + t] * v;
    }
  }
#pragma unroll
  for (int c = 0; c < NCH; ++c) {
    float v = acc[c] + bl[c];
    v = v > 0.f ? v : 0.f;
    act1[(size_t)idx * NCH + c] = (_Float16)v;
  }
}

// ---------------------------------------------------------------------------
// conv2: 16 -> 16 channels, implicit GEMM on v_wmma_f32_16x16x32_f16.
// One wave = TWO 16-row tiles sharing one B fragment (two independent WMMA
// chains).  Branchless halo: an 82-bit per-row tap-validity bitmask masks
// the *address* to a zeroed null page.  Tap deltas from a broadcast LDS LUT.
// Ping-pong double-buffered pipeline (unroll x2) => no cur/nxt reg copies.
// ---------------------------------------------------------------------------
struct Frag { v16h a0, a1, bf; };

__global__ void nc_conv2_wmma(const char* __restrict__ wsBase,   // d_ws
                              const float* __restrict__ w2,
                              const float* __restrict__ b2,
                              _Float16* __restrict__ act2) {
  __shared__ __align__(32) _Float16 wl[(NTAP + 1) * NCH * NCH];  // 41,984 B
  __shared__ __align__(8) int dlut[NTAP + 1];                    // byte deltas
  for (int e = threadIdx.x; e < (NTAP + 1) * 256; e += blockDim.x) {
    int t = e >> 8, r = e & 255, o = r >> 4, ci = r & 15;
    wl[e] = (t < NTAP) ? (_Float16)w2[(o * NCH + ci) * NTAP + t] : (_Float16)0.f;
  }
  if (threadIdx.x < NTAP + 1) {
    int t = threadIdx.x;
    int k4 = t % 3, q = t / 3; int k3 = q % 3; q /= 3; int k2 = q % 3, k1 = q / 3;
    int d = ((k1 - 1) * 15625 + (k2 - 1) * 625 + (k3 - 1) * 25 + (k4 - 1)) * (NCH * 2);
    dlut[t] = (t < NTAP) ? d : 0;
  }
  __syncthreads();

  const int lane = threadIdx.x & 31;
  const int wave = threadIdx.x >> 5;
  const int l15  = lane & 15;                 // A row M, C column N
  const int lsel = lane >> 4;                 // 0: K 0-7,16-23 (cin 0-7), 1: K 8-15,24-31 (cin 8-15)
  const int co   = lsel * 8;                  // cin sub-range for A fragment
  const int tileBase = (blockIdx.x * 8 + wave) * 32;

  // Per-tile-row: byte offset of row center + 82-bit tap-validity bitmask.
  unsigned offR[2];
  unsigned long long mLo[2];
  unsigned mHi[2];
#pragma unroll
  for (int s = 0; s < 2; ++s) {
    int m = tileBase + 16 * s + l15;
    int mm = (m < TOT) ? m : 0;
    int f = mm % SP;
    int i4 = f % DIM;  int q = f / DIM;
    int i3 = q % DIM;  q /= DIM;
    int i2 = q % DIM;  int i1 = q / DIM;
    offR[s] = GUARD + (unsigned)m * (NCH * 2u) + (unsigned)co * 2u;

    unsigned a1m = 0, a2m = 0, a3m = 0, a4m = 0;
#pragma unroll
    for (int k = 0; k < 3; ++k) {
      a1m |= ((unsigned)(i1 + k - 1) < DIM) ? (1u << k) : 0u;
      a2m |= ((unsigned)(i2 + k - 1) < DIM) ? (1u << k) : 0u;
      a3m |= ((unsigned)(i3 + k - 1) < DIM) ? (1u << k) : 0u;
      a4m |= ((unsigned)(i4 + k - 1) < DIM) ? (1u << k) : 0u;
    }
    // bit t (tap) = a4m[t%3] & a3m[(t/3)%3] & a2m[(t/9)%3] & a1m[t/27]
    unsigned long long p4 = (unsigned long long)a4m * 0x9249249249249249ULL;
    unsigned h4 = (((a4m >> 1) | (a4m << 2)) & 7u) * 0x249249u;        // (u+64)%3=(u+1)%3
    unsigned e3 = ((a3m & 1u) * 0x7u) | (((a3m >> 1) & 1u) * 0x38u)
                | (((a3m >> 2) & 1u) * 0x1C0u);
    unsigned long long p3 = (unsigned long long)e3 * 0x8040201008040201ULL;
    unsigned e3r = ((e3 >> 1) | (e3 << 8)) & 0x1FFu;                   // rotr9 by 1
    unsigned h3 = e3r | (e3r << 9);
    unsigned e2 = ((a2m & 1u) * 0x1FFu) | (((a2m >> 1) & 1u) * (0x1FFu << 9))
                | (((a2m >> 2) & 1u) * (0x1FFu << 18));
    unsigned long long p2 = (unsigned long long)e2 * 0x40000008000001ULL;
    unsigned h2 = ((e2 >> 10) | (e2 << 17)) & 0x7FFFFFFu;              // rotr27 by 10
    unsigned long long p1 = ((a1m & 1u) ? 0x7FFFFFFULL : 0)
                          | ((a1m & 2u) ? (0x7FFFFFFULL << 27) : 0)
                          | ((a1m & 4u) ? (0x3FFULL << 54) : 0);
    unsigned h1 = (a1m & 4u) ? 0x3FFFFu : 0u;
    unsigned long long lo = p4 & p3 & p2 & p1;
    unsigned hi = h4 & h3 & h2 & h1 & 0x1FFFFu;                        // tap 81 = 0
    mLo[s] = (m < TOT) ? lo : 0ull;
    mHi[s] = (m < TOT) ? hi : 0u;
  }
  const int bfBase = lsel * 8 + l15 * 16;     // halfs; +512 per iteration

  auto fetch = [&](int tp) -> Frag {
    Frag fr;
    int2 dd = *(const int2*)&dlut[2 * tp];    // broadcast ds_load_b64
#pragma unroll
    for (int s = 0; s < 2; ++s) {
      unsigned b0 = (unsigned)mLo[s] & 1u;
      unsigned b1 = ((unsigned)mLo[s] >> 1) & 1u;
      unsigned o0 = (offR[s] + (unsigned)dd.x) & (0u - b0);   // masked -> null page
      unsigned o1 = (offR[s] + (unsigned)dd.y) & (0u - b1);
      mLo[s] = (mLo[s] >> 2) | ((unsigned long long)(mHi[s] & 3u) << 62);
      mHi[s] >>= 2;
      v8h r0 = *(const v8h*)(wsBase + o0);
      v8h r1 = *(const v8h*)(wsBase + o1);
      v16h a = __builtin_shufflevector(r0, r1, 0, 1, 2, 3, 4, 5, 6, 7,
                                               8, 9, 10, 11, 12, 13, 14, 15);
      if (s == 0) fr.a0 = a; else fr.a1 = a;
    }
    // B: elems 0-7 = tap 2tp cin 8*lsel..+7, elems 8-15 = tap 2tp+1 same cins;
    // at tp==40 the second half lands exactly on the zeroed tap-81 slot.
    v8h b0v = *(const v8h*)&wl[bfBase + tp * 512];
    v8h b1v = *(const v8h*)&wl[bfBase + tp * 512 + 256];
    fr.bf = __builtin_shufflevector(b0v, b1v, 0, 1, 2, 3, 4, 5, 6, 7,
                                    8, 9, 10, 11, 12, 13, 14, 15);
    return fr;
  };

  v8f c0 = {}, c1 = {};
  // Ping-pong pipeline: fetch order is strictly 0,1,2,...,40 (mask shift!).
  Frag fA = fetch(0);
#pragma unroll 1
  for (int tp = 0; tp < 40; tp += 2) {
    Frag fB = fetch(tp + 1);
    c0 = __builtin_amdgcn_wmma_f32_16x16x32_f16(false, fA.a0, false, fA.bf,
                                                (short)0, c0, false, false);
    c1 = __builtin_amdgcn_wmma_f32_16x16x32_f16(false, fA.a1, false, fA.bf,
                                                (short)0, c1, false, false);
    fA = fetch(tp + 2);                       // preloads pair tp+2 (<= 40)
    c0 = __builtin_amdgcn_wmma_f32_16x16x32_f16(false, fB.a0, false, fB.bf,
                                                (short)0, c0, false, false);
    c1 = __builtin_amdgcn_wmma_f32_16x16x32_f16(false, fB.a1, false, fB.bf,
                                                (short)0, c1, false, false);
  }
  c0 = __builtin_amdgcn_wmma_f32_16x16x32_f16(false, fA.a0, false, fA.bf,
                                              (short)0, c0, false, false);
  c1 = __builtin_amdgcn_wmma_f32_16x16x32_f16(false, fA.a1, false, fA.bf,
                                              (short)0, c1, false, false);

  float bias = b2[l15];
#pragma unroll
  for (int r = 0; r < 8; ++r) {               // C: VGPR r -> M = r + 8*lsel, N = l15
    int p0 = tileBase + r + 8 * lsel;
    int p1 = p0 + 16;
    if (p0 < TOT) {
      float v = c0[r] + bias; v = v > 0.f ? v : 0.f;
      act2[(size_t)p0 * NCH + l15] = (_Float16)v;
    }
    if (p1 < TOT) {
      float v = c1[r] + bias; v = v > 0.f ? v : 0.f;
      act2[(size_t)p1 * NCH + l15] = (_Float16)v;
    }
  }
}

// ---------------------------------------------------------------------------
// conv3: 16 -> 1 channel, f32 VALU over f16 activations.
// perm==0: out = relu(.)        perm!=0: out[transposed idx] += relu(.)
// ---------------------------------------------------------------------------
__global__ void nc_conv3(const _Float16* __restrict__ act2,
                         const float* __restrict__ w3, const float* __restrict__ b3,
                         float* __restrict__ out, int perm) {
  __shared__ float wl[NTAP * NCH];            // [tap][cin]
  for (int e = threadIdx.x; e < NTAP * NCH; e += blockDim.x) {
    int t = e >> 4, ci = e & 15;
    wl[e] = w3[ci * NTAP + t];
  }
  __syncthreads();

  int idx = blockIdx.x * blockDim.x + threadIdx.x;
  if (idx >= TOT) return;
  int b = idx / SP, f = idx - b * SP;
  int i4 = f % DIM;  int q = f / DIM;
  int i3 = q % DIM;  q /= DIM;
  int i2 = q % DIM;  int i1 = q / DIM;

  float acc = 0.f;
  for (int t = 0; t < NTAP; ++t) {
    int k4 = t % 3, r = t / 3;
    int k3 = r % 3; r /= 3;
    int k2 = r % 3; int k1 = r / 3;
    int j1 = i1 + k1 - 1, j2 = i2 + k2 - 1, j3 = i3 + k3 - 1, j4 = i4 + k4 - 1;
    if ((unsigned)j1 < DIM && (unsigned)j2 < DIM &&
        (unsigned)j3 < DIM && (unsigned)j4 < DIM) {
      int src = b * SP + (((j1 * DIM + j2) * DIM + j3) * DIM + j4);
      const v8h* p = (const v8h*)(act2 + (size_t)src * NCH);
      v8h p0 = p[0], p1 = p[1];
#pragma unroll
      for (int i = 0; i < 8; ++i) {
        acc += (float)p0[i] * wl[t * NCH + i];
        acc += (float)p1[i] * wl[t * NCH + 8 + i];
      }
    }
  }
  acc += b3[0];
  acc = acc > 0.f ? acc : 0.f;

  if (perm) {
    int of = ((i3 * DIM + i4) * DIM + i1) * DIM + i2;   // write transposed
    out[(size_t)b * SP + of] += acc;
  } else {
    out[idx] = acc;
  }
}

// ---------------------------------------------------------------------------
extern "C" void kernel_launch(void* const* d_in, const int* in_sizes, int n_in,
                              void* d_out, int out_size, void* d_ws, size_t ws_size,
                              hipStream_t stream) {
  const float* x  = (const float*)d_in[0];
  const float* w1 = (const float*)d_in[1];
  const float* b1 = (const float*)d_in[2];
  const float* w2 = (const float*)d_in[3];
  const float* b2 = (const float*)d_in[4];
  const float* w3 = (const float*)d_in[5];
  const float* b3 = (const float*)d_in[6];
  float* out = (float*)d_out;

  // d_ws layout: [null page + 1MB guard | act1 (100MB f16) | act2 (100MB f16) | 1MB guard]
  _Float16* act1 = (_Float16*)((char*)d_ws + GUARD);
  _Float16* act2 = act1 + (size_t)TOT * NCH;

  dim3 blk(256);
  dim3 grdP((TOT + 255) / 256);               // pointwise kernels
  dim3 grdW((TOT + 255) / 256);               // 8 waves x 2 tiles x 16 rows = 256

  nc_zero<<<1, 64, 0, stream>>>((float*)d_ws);
  for (int perm = 0; perm < 2; ++perm) {
    nc_conv1<<<grdP, blk, 0, stream>>>(x, w1, b1, act1, perm);
    nc_conv2_wmma<<<grdW, blk, 0, stream>>>((const char*)d_ws, w2, b2, act2);
    nc_conv3<<<grdP, blk, 0, stream>>>(act2, w3, b3, out, perm);
  }
}
